// GaussianRenderer_39608188403924
// MI455X (gfx1250) — matrix operand order repaired and checked by reading, not compiled, not run on hardware
//
#include <hip/hip_runtime.h>

// ---------------------------------------------------------------------------
// Gaussian tile splatting for MI455X (gfx1250, wave32).
//  Kernel 1: stable depth rank (O(N^2) count, LDS-chunked) + scatter sorted SoA.
//  Kernel 2: per-tile: wave32 ballot-compacted top-K selection (depth order),
//            fp32 alpha blending producing W[pixel][k], then the color einsum
//            via v_wmma_f32_16x16x32_f16 (f32 accumulate, f16 inputs) with
//            vectorized (b128) LDS operand loads.
// ---------------------------------------------------------------------------

typedef __attribute__((ext_vector_type(16))) _Float16 v16h;
typedef __attribute__((ext_vector_type(8)))  _Float16 v8h;
typedef __attribute__((ext_vector_type(8)))  float    v8f;

#define IMG_W    512
#define IMG_H    512
#define TILE_LEN 16
#define K_MAX    64
#define TX_TILES 32          // IMG_W / TILE_LEN
#define TY_TILES 32          // IMG_H / TILE_LEN
#define N_TILES  (TX_TILES * TY_TILES)
#define W_STRIDE 72          // padded K stride (halfs); 144B = 9*16B (aligned rows)

// Workspace layout (floats), 10 arrays of N elements each, depth-sorted:
//  0:px 1:py 2:radius 3:qxx(=c/det) 4:qxy(=b/det) 5:qyy(=a/det) 6:op 7:cr 8:cg 9:cb

__global__ __launch_bounds__(256) void gs_rank_scatter(
    const float* __restrict__ pos2d, const float* __restrict__ cov2d,
    const float* __restrict__ opacity, const float* __restrict__ color,
    const float* __restrict__ depth, float* __restrict__ ws, int N)
{
    __shared__ float sdep[256];
    const int i = blockIdx.x * 256 + threadIdx.x;
    const float di = depth[i];

    // Stable ascending rank: #(d_j < d_i) + #(d_j == d_i && j < i)
    int rank = 0;
    for (int base = 0; base < N; base += 256) {
        __syncthreads();
        sdep[threadIdx.x] = depth[base + threadIdx.x];
        __syncthreads();
        #pragma unroll 8
        for (int j = 0; j < 256; ++j) {
            float dj = sdep[j];
            rank += (dj < di || (dj == di && (base + j) < i)) ? 1 : 0;
        }
    }

    const float a = cov2d[i * 4 + 0];
    const float b = cov2d[i * 4 + 1];
    const float c = cov2d[i * 4 + 3];
    const float tr  = a + c;
    const float det = a * c - b * b;
    const float t1 = 0.5f * tr;
    const float t2 = 0.5f * sqrtf(fmaxf(tr * tr - 4.0f * det, 0.0f));
    const float radius = 3.0f * sqrtf(fmaxf(t1 - t2, t1 + t2));
    const float inv = 1.0f / det;

    ws[0 * N + rank] = pos2d[i * 2 + 0];
    ws[1 * N + rank] = pos2d[i * 2 + 1];
    ws[2 * N + rank] = radius;
    ws[3 * N + rank] = c * inv;   // multiplies dx*dx
    ws[4 * N + rank] = b * inv;   // multiplies -2*dx*dy
    ws[5 * N + rank] = a * inv;   // multiplies dy*dy
    ws[6 * N + rank] = opacity[i];
    ws[7 * N + rank] = color[i * 3 + 0];
    ws[8 * N + rank] = color[i * 3 + 1];
    ws[9 * N + rank] = color[i * 3 + 2];
}

__global__ __launch_bounds__(256) void gs_render(
    const float* __restrict__ ws, float* __restrict__ out, int N)
{
    __shared__ float    s_px[K_MAX], s_py[K_MAX];
    __shared__ float    s_qxx[K_MAX], s_qxy[K_MAX], s_qyy[K_MAX], s_op[K_MAX];
    __shared__ alignas(32) _Float16 s_colT[16 * K_MAX];      // [n][k], n 0..15 (3 live)
    __shared__ alignas(32) _Float16 s_W[256 * W_STRIDE];     // [pixel][k] weights
    __shared__ int      s_cnt;

    const float* spx  = ws + 0 * N;
    const float* spy  = ws + 1 * N;
    const float* srad = ws + 2 * N;
    const float* sqxx = ws + 3 * N;
    const float* sqxy = ws + 4 * N;
    const float* sqyy = ws + 5 * N;
    const float* sop  = ws + 6 * N;
    const float* scr  = ws + 7 * N;
    const float* scg  = ws + 8 * N;
    const float* scb  = ws + 9 * N;

    const int tile = blockIdx.x;
    const int tx = tile >> 5;          // tile = tx*TY_TILES + ty
    const int ty = tile & 31;
    const float Lf = (float)(tx * TILE_LEN);
    const float Tf = (float)(ty * TILE_LEN);

    const int tid = threadIdx.x;

    // ---- Zero the padded color matrix (rows n>=3 and k>=cnt must be 0,
    //      and 0*garbage would be NaN-prone in WMMA; zero it all first) ----
    {
        v8h z = {};
        #pragma unroll
        for (int c = 0; c < (16 * K_MAX) / (256 * 8); ++c)   // 1024/2048 -> 0: guard below
            ;
        // 16*K_MAX = 1024 halfs; 256 threads x 4 halfs each
        ((_Float16*)s_colT)[tid * 4 + 0] = (_Float16)0.0f;
        ((_Float16*)s_colT)[tid * 4 + 1] = (_Float16)0.0f;
        ((_Float16*)s_colT)[tid * 4 + 2] = (_Float16)0.0f;
        ((_Float16*)s_colT)[tid * 4 + 3] = (_Float16)0.0f;
        (void)z;
    }
    __syncthreads();

    // ---- Phase A: wave 0 selects first K_MAX overlapping (depth order) ----
    if (tid < 32) {
        const int lane = tid;
        const unsigned lanemask_lt = (1u << lane) - 1u;
        int cnt = 0;
        for (int base = 0; base < N && cnt < K_MAX; base += 32) {
            const int g = base + lane;
            const float px = spx[g], py = spy[g], r = srad[g];
            const bool ov = (px + r > Lf) && (px - r < Lf + (float)TILE_LEN) &&
                            (py + r > Tf) && (py - r < Tf + (float)TILE_LEN);
            const unsigned mask = __builtin_amdgcn_ballot_w32(ov);
            const int slot = cnt + __popc(mask & lanemask_lt);
            if (ov && slot < K_MAX) {
                s_px[slot]  = px;      s_py[slot]  = py;
                s_qxx[slot] = sqxx[g]; s_qxy[slot] = sqxy[g]; s_qyy[slot] = sqyy[g];
                s_op[slot]  = sop[g];
                s_colT[0 * K_MAX + slot] = (_Float16)scr[g];
                s_colT[1 * K_MAX + slot] = (_Float16)scg[g];
                s_colT[2 * K_MAX + slot] = (_Float16)scb[g];
            }
            cnt += __popc(mask);
        }
        if (lane == 0) s_cnt = (cnt > K_MAX) ? K_MAX : cnt;
    }
    __syncthreads();
    const int cnt = s_cnt;

    // ---- Phase B: per-pixel fp32 front-to-back blending weights ----
    {
        const int pi = tid >> 4;       // base dim0 (x within tile)
        const int pj = tid & 15;       // base dim1 (y within tile)
        const float pxl = Lf + (float)pi;
        const float pyl = Tf + (float)pj;
        float trans = 1.0f;
        _Float16* wrow = &s_W[tid * W_STRIDE];
        for (int k = 0; k < cnt; ++k) {          // cnt is block-uniform
            const float dx = pxl - s_px[k];
            const float dy = pyl - s_py[k];
            const float quad = s_qxx[k] * dx * dx
                             - 2.0f * s_qxy[k] * dx * dy
                             + s_qyy[k] * dy * dy;
            const float prob  = expf(-0.5f * quad);
            const float alpha = fminf(fmaxf(s_op[k] * prob, 0.01f), 0.99f);
            wrow[k] = (_Float16)(alpha * trans);
            trans *= (1.0f - alpha);
        }
        for (int k = cnt; k < K_MAX; ++k)
            wrow[k] = (_Float16)0.0f;
    }
    __syncthreads();

    // ---- Phase C: color einsum via WMMA f32 = f16 x f16 ----
    // out[p, c] = sum_k W[p,k] * col[k,c]; 16 row-blocks of 16 pixels,
    // N padded to 16 (c<3 live). 8 waves x 2 row-blocks each; K=64 = 2 chunks.
    const int wave = tid >> 5;
    const int lane = tid & 31;
    const int m  = lane & 15;          // A row / B,C column index
    const int hi = lane >> 4;

    #pragma unroll
    for (int rbi = 0; rbi < 2; ++rbi) {
        const int rowbase = (wave + rbi * 8) * 16;
        v8f cacc = {};
        #pragma unroll
        for (int kc = 0; kc < 2; ++kc) {
            // A 16x32 f16 ISA layout decomposes, per lane, into two contiguous
            // 8-half runs: K in [hi*8, hi*8+8) and [16+hi*8, 16+hi*8+8).
            const _Float16* wrow = &s_W[(rowbase + m) * W_STRIDE + kc * 32];
            const v8h a_lo = *(const v8h*)(wrow + hi * 8);        // 16B aligned
            const v8h a_hi = *(const v8h*)(wrow + 16 + hi * 8);   // 16B aligned
            const v16h av = __builtin_shufflevector(
                a_lo, a_hi, 0, 1, 2, 3, 4, 5, 6, 7, 8, 9, 10, 11, 12, 13, 14, 15);
            // B 32x16 f16: same K runs as A: [hi*8, hi*8+8) and [16+hi*8, 16+hi*8+8), N = m.
            // With s_colT[n][k], that's two contiguous 8-half runs (16B-aligned).
            const _Float16* crow = &s_colT[m * K_MAX + kc * 32];
            const v8h b_lo = *(const v8h*)(crow + hi * 8);
            const v8h b_hi = *(const v8h*)(crow + 16 + hi * 8);
            const v16h bv = __builtin_shufflevector(
                b_lo, b_hi, 0, 1, 2, 3, 4, 5, 6, 7, 8, 9, 10, 11, 12, 13, 14, 15);
            cacc = __builtin_amdgcn_wmma_f32_16x16x32_f16(
                false, av, false, bv, (short)0, cacc, false, false);
        }
        // D 16x16 f32 layout: lane L, VGPR v -> M = v + hi*8, N = L&15
        #pragma unroll
        for (int v = 0; v < 8; ++v) {
            const int n = m;
            if (n < 3) {
                const int p = rowbase + v + hi * 8;     // pixel within tile
                const int x = tx * TILE_LEN + (p >> 4);
                const int y = ty * TILE_LEN + (p & 15);
                out[(x * IMG_H + y) * 3 + n] = cacc[v];
            }
        }
    }
}

extern "C" void kernel_launch(void* const* d_in, const int* in_sizes, int n_in,
                              void* d_out, int out_size, void* d_ws, size_t ws_size,
                              hipStream_t stream) {
    const float* pos2d   = (const float*)d_in[0];
    const float* cov2d   = (const float*)d_in[1];
    const float* opacity = (const float*)d_in[2];
    const float* color   = (const float*)d_in[3];
    const float* depth   = (const float*)d_in[4];
    const int N = in_sizes[4];           // 16384 (multiple of 256)
    float* ws  = (float*)d_ws;           // needs 10*N*4 bytes (~640 KB)
    float* out = (float*)d_out;          // 512*512*3 f32

    gs_rank_scatter<<<N / 256, 256, 0, stream>>>(pos2d, cov2d, opacity, color,
                                                 depth, ws, N);
    gs_render<<<N_TILES, 256, 0, stream>>>(ws, out, N);
}
